// TreeVAE_41266045780410
// MI455X (gfx1250) — matrix operand order repaired and checked by reading, not compiled, not run on hardware
//
#include <hip/hip_runtime.h>

// ---------------------------------------------------------------------------
// Types / helpers
// ---------------------------------------------------------------------------
typedef __attribute__((ext_vector_type(16))) __bf16 v16bf;
typedef __attribute__((ext_vector_type(8)))  __bf16 v8bf;
typedef __attribute__((ext_vector_type(8)))  float  v8f;

__device__ __forceinline__ __bf16 f2bf(float x) {
  unsigned u = __builtin_bit_cast(unsigned, x);
  unsigned r = (u + 0x7FFFu + ((u >> 16) & 1u)) >> 16;   // round-to-nearest-even
  return __builtin_bit_cast(__bf16, (unsigned short)r);
}

// ---------------------------------------------------------------------------
// Elementwise cast kernels
// ---------------------------------------------------------------------------
__global__ void k_cast_bf16(const float* __restrict__ in, __bf16* __restrict__ out, long n) {
  long i = (long)blockIdx.x * blockDim.x + threadIdx.x;
  long st = (long)gridDim.x * blockDim.x;
  for (; i < n; i += st) out[i] = f2bf(in[i]);
}

__global__ void k_cast_relu_bf16(const float* __restrict__ in, __bf16* __restrict__ out, long n) {
  long i = (long)blockIdx.x * blockDim.x + threadIdx.x;
  long st = (long)gridDim.x * blockDim.x;
  for (; i < n; i += st) out[i] = f2bf(fmaxf(in[i], 0.f));
}

// in: [K][N] f32 (row-major)  ->  out: [N][K] bf16
__global__ void k_cast_tr_bf16(const float* __restrict__ in, __bf16* __restrict__ out,
                               int K, int N) {
  long n = (long)K * N;
  long i = (long)blockIdx.x * blockDim.x + threadIdx.x;
  long st = (long)gridDim.x * blockDim.x;
  for (; i < n; i += st) {
    int k = (int)(i / N), c = (int)(i % N);
    out[(long)c * K + k] = f2bf(in[i]);
  }
}

// dw1 [64][128][4][4] -> 4 parity weight mats [par][o=64][c*4 + ky'*2 + kx'] bf16
__global__ void k_dw1_par_cast(const float* __restrict__ w, __bf16* __restrict__ out) {
  int i = blockIdx.x * blockDim.x + threadIdx.x;
  if (i >= 4 * 64 * 512) return;
  int t = i & 3;             // ky'*2 + kx'
  int c = (i >> 2) & 127;    // in channel
  int o = (i >> 9) & 63;     // out channel
  int par = i >> 15;         // py*2 + px
  int py = par >> 1, px = par & 1;
  int ky = py + 2 * (t >> 1), kx = px + 2 * (t & 1);
  out[i] = f2bf(w[((o * 128 + c) * 4 + ky) * 4 + kx]);
}

// ---------------------------------------------------------------------------
// Generic im2col (f32 NCHW -> bf16 patch matrix [B*OH*OW][C*KH*KW]),
// optional ReLU folded on the read side.
// ---------------------------------------------------------------------------
__global__ void k_im2col(const float* __restrict__ in, __bf16* __restrict__ col,
                         int C, int H, int W, int OH, int OW,
                         int KH, int KW, int stride, int pady, int padx,
                         int relu, long total) {
  long i = (long)blockIdx.x * blockDim.x + threadIdx.x;
  long st = (long)gridDim.x * blockDim.x;
  const int K = C * KH * KW;
  for (; i < total; i += st) {
    int k = (int)(i % K);
    long row = i / K;
    int p = (int)(row % (OH * OW));
    int b = (int)(row / (OH * OW));
    int oy = p / OW, ox = p % OW;
    int c = k / (KH * KW);
    int t = k % (KH * KW);
    int ky = t / KW, kx = t % KW;
    int iy = oy * stride - pady + ky;
    int ix = ox * stride - padx + kx;
    float v = 0.f;
    if ((unsigned)iy < (unsigned)H && (unsigned)ix < (unsigned)W)
      v = in[(((long)b * C + c) * H + iy) * W + ix];
    if (relu) v = fmaxf(v, 0.f);
    col[i] = f2bf(v);
  }
}

// ---------------------------------------------------------------------------
// WMMA bf16 GEMM:  out = act( A[M,K] * Wt[N,K]^T + bias (+ resid) )
//
// Block = 128 threads (4 waves) computes a 64(M) x 64(N) macro-tile.
//   Wave w owns m-tile (bm*64 + w*16) and all four 16-wide n-tiles:
//   one A fragment is reused across 4 v_wmma_f32_16x16x32_bf16 per K-step.
//   The shared 64x32 bf16 B tile is staged to LDS with
//   global_load_async_to_lds_b128 (ASYNCcnt), double-buffered:
//     stage(buf^1) -> 4 WMMAs from LDS buf -> s_wait_asynccnt 0 -> barrier.
// Fragment layouts per CDNA5 ISA 7.12.2 (wave32):
//   A 16x32: r=lane&15 row; h=lane>>4; 16 bf16 = K{h*8..+7} ++ K{16+h*8..+7}
//   B 32x16: c=lane&15 col; 16 bf16 = K{h*8..+7} ++ K{16+h*8..+7}  (Wt stored [N][K])
//   D: m = v + 8*h, n = lane&15 across 8 f32 VGPRs
// P==0 -> dense row-major store; P>0 -> conv NCHW scatter (see round 1).
// ---------------------------------------------------------------------------
__global__ __launch_bounds__(128)
void k_gemm_bf16_wmma(const __bf16* __restrict__ A, const __bf16* __restrict__ Wt,
                      const float* __restrict__ bias, const float* __restrict__ resid,
                      float* __restrict__ out,
                      int M, int N, int K,
                      int P, int WA, int ys, int xs, int istr, int relu) {
  __shared__ __align__(16) __bf16 btile[2][64][32];   // 8 KB double-buffered B tile

  const int tid  = threadIdx.x;
  const int wave = tid >> 5;
  const int lane = tid & 31;
  const int nb   = N >> 6;
  const int bm   = (int)blockIdx.x / nb;
  const int bn   = (int)blockIdx.x % nb;
  const int r = lane & 15, hh = lane >> 4;

  // B staging: 64 rows x 64B = 256 chunks of 16B; 2 chunks per thread.
  const int ch0 = tid, ch1 = tid + 128;
  const int row0 = ch0 >> 2, q0 = ch0 & 3;
  const int row1 = ch1 >> 2, q1 = ch1 & 3;
  const __bf16* Bg0 = Wt + (size_t)(bn * 64 + row0) * K + q0 * 8;
  const __bf16* Bg1 = Wt + (size_t)(bn * 64 + row1) * K + q1 * 8;

  auto stage = [&](int k0, int buf) {
    unsigned l0 = (unsigned)(unsigned long long)(void*)&btile[buf][row0][q0 * 8];
    unsigned l1 = (unsigned)(unsigned long long)(void*)&btile[buf][row1][q1 * 8];
    const __bf16* g0 = Bg0 + k0;
    const __bf16* g1 = Bg1 + k0;
    asm volatile("global_load_async_to_lds_b128 %0, %1, off"
                 :: "v"(l0), "v"(g0) : "memory");
    asm volatile("global_load_async_to_lds_b128 %0, %1, off"
                 :: "v"(l1), "v"(g1) : "memory");
  };

  const __bf16* Ap = A + (size_t)(bm * 64 + wave * 16 + r) * K + hh * 8;

  v8f acc[4];
#pragma unroll
  for (int i = 0; i < 4; ++i) acc[i] = (v8f){0.f, 0.f, 0.f, 0.f, 0.f, 0.f, 0.f, 0.f};

  stage(0, 0);
  asm volatile("s_wait_asynccnt 0x0" ::: "memory");
  __syncthreads();

  const int nk = K >> 5;
  for (int t = 0; t < nk; ++t) {
    const int k0 = t << 5;
    const int buf = t & 1;
    if (t + 1 < nk) stage(k0 + 32, buf ^ 1);

    __builtin_prefetch(Ap + k0 + 64, 0, 1);
    v8bf a0 = *(const v8bf*)(Ap + k0);
    v8bf a1 = *(const v8bf*)(Ap + k0 + 16);
    v16bf av = __builtin_shufflevector(a0, a1, 0,1,2,3,4,5,6,7,8,9,10,11,12,13,14,15);

#pragma unroll
    for (int ntl = 0; ntl < 4; ++ntl) {
      const __bf16* bp = &btile[buf][ntl * 16 + r][hh * 8];
      v8bf b0 = *(const v8bf*)bp;
      v8bf b1 = *(const v8bf*)(bp + 16);
      v16bf bv = __builtin_shufflevector(b0, b1, 0,1,2,3,4,5,6,7,8,9,10,11,12,13,14,15);
      acc[ntl] = __builtin_amdgcn_wmma_f32_16x16x32_bf16(false, av, false, bv,
                                                         (short)0, acc[ntl], false, false);
    }

    asm volatile("s_wait_asynccnt 0x0" ::: "memory");
    __syncthreads();
  }

  const int mrow = bm * 64 + wave * 16;
#pragma unroll
  for (int ntl = 0; ntl < 4; ++ntl) {
    const int n = bn * 64 + ntl * 16 + r;
    const float bb = bias ? bias[n] : 0.f;
#pragma unroll
    for (int v = 0; v < 8; ++v) {
      const int m = mrow + v + 8 * hh;
      size_t idx;
      if (P > 0) {
        const int b = m / P, p = m - b * P;
        const int Y = p / WA, X = p - Y * WA;
        idx = (size_t)(b * N + n) * istr + (size_t)Y * ys + (size_t)X * xs;
      } else {
        idx = (size_t)m * N + n;
      }
      float val = acc[ntl][v] + bb;
      if (resid) val += resid[idx];
      if (relu)  val = fmaxf(val, 0.f);
      out[idx] = val;
    }
  }
}

// ---------------------------------------------------------------------------
// conv1: 3->64, 4x4 s2 p1, ReLU (direct; tiny K so VALU is right)
// ---------------------------------------------------------------------------
__global__ void k_conv1(const float* __restrict__ x, const float* __restrict__ w,
                        const float* __restrict__ bias, float* __restrict__ out) {
  int i = blockIdx.x * blockDim.x + threadIdx.x;
  if (i >= 64 * 64 * 32 * 32) return;
  int xw = i & 31, yy = (i >> 5) & 31, o = (i >> 10) & 63, b = i >> 16;
  float s = bias[o];
  for (int c = 0; c < 3; ++c)
    for (int ky = 0; ky < 4; ++ky) {
      int iy = 2 * yy - 1 + ky;
      if ((unsigned)iy >= 64u) continue;
      for (int kx = 0; kx < 4; ++kx) {
        int ix = 2 * xw - 1 + kx;
        if ((unsigned)ix >= 64u) continue;
        s += x[((b * 3 + c) * 64 + iy) * 64 + ix] * w[((o * 3 + c) * 4 + ky) * 4 + kx];
      }
    }
  out[i] = fmaxf(s, 0.f);
}

// ---------------------------------------------------------------------------
// convT2: 64->3, 4x4 s2 SAME transpose conv (direct, 3 out channels)
// ---------------------------------------------------------------------------
__global__ void k_convT2(const float* __restrict__ gup, const float* __restrict__ w,
                         const float* __restrict__ bias, float* __restrict__ out) {
  int i = blockIdx.x * blockDim.x + threadIdx.x;
  if (i >= 64 * 3 * 64 * 64) return;
  int p = i;
  int xw = p & 63; p >>= 6;
  int yy = p & 63; p >>= 6;
  int o = p % 3;
  int b = p / 3;
  int py = yy & 1, px = xw & 1;
  float s = bias[o];
  for (int ty = 0; ty < 2; ++ty) {
    int ky = py + 2 * ty;
    int uy = yy + ky - 2;
    if (uy < 0) continue;
    uy >>= 1;
    if (uy >= 32) continue;
    for (int tx = 0; tx < 2; ++tx) {
      int kx = px + 2 * tx;
      int ux = xw + kx - 2;
      if (ux < 0) continue;
      ux >>= 1;
      if (ux >= 32) continue;
      for (int c = 0; c < 64; ++c)
        s += gup[((b * 64 + c) * 32 + uy) * 32 + ux] * w[((o * 64 + c) * 4 + ky) * 4 + kx];
    }
  }
  out[i] = s;
}

// ---------------------------------------------------------------------------
// Tree reparameterization (chain tree, gamma = 0.5)
// ---------------------------------------------------------------------------
__global__ void k_reparam(const float* __restrict__ mu, const float* __restrict__ lv,
                          const float* __restrict__ e,
                          float* __restrict__ z, __bf16* __restrict__ zbf) {
  int i = blockIdx.x * blockDim.x + threadIdx.x;
  if (i >= 64 * 256) return;
  int n = i >> 8, d = i & 255;
  float s = 0.f;
  if (n == 0) {
    s = e[d];
  } else {
    for (int a = 0; a < n; ++a) {
      float c = exp2f((float)(a - n));                 // 0.5^(n-a)
      c *= (a == 0) ? 1.f : 0.8660254037844386f;       // sqrt(1-0.25)
      s += c * e[a * 256 + d];
    }
  }
  float sg = expf(0.5f * lv[i]);
  float zv = mu[i] + sg * s;
  z[i] = zv;
  zbf[i] = f2bf(zv);
}

// ---------------------------------------------------------------------------
// Loss reductions (deterministic: block-tree + fixed-order final pass)
// ---------------------------------------------------------------------------
__global__ void k_sse_partial(const float* __restrict__ recon, const float* __restrict__ x,
                              float* __restrict__ part, long n) {
  __shared__ float sm[256];
  float s = 0.f;
  for (long i = (long)blockIdx.x * 256 + threadIdx.x; i < n; i += (long)gridDim.x * 256) {
    float d = recon[i] - x[i];
    s += d * d;
  }
  sm[threadIdx.x] = s;
  __syncthreads();
  for (int st = 128; st > 0; st >>= 1) {
    if ((int)threadIdx.x < st) sm[threadIdx.x] += sm[threadIdx.x + st];
    __syncthreads();
  }
  if (threadIdx.x == 0) part[blockIdx.x] = sm[0];
}

__global__ void k_finalize(const float* __restrict__ mu, const float* __restrict__ lv,
                           const float* __restrict__ z,
                           const float* __restrict__ part, int npart,
                           float* __restrict__ out3) {
  __shared__ float s_z2[256], s_ent[256];
  __shared__ int s_act[256];
  int d = threadIdx.x;
  float z2 = 0.f, ent = 0.f, kls = 0.f;
  for (int b = 0; b < 64; ++b) {
    int i = b * 256 + d;
    float m = mu[i], l = lv[i], zz = z[i];
    float s2 = expf(l);
    float lg = logf(s2 + 1e-8f);
    z2 += zz * zz;
    ent += 1.f + lg;
    kls += -0.5f * (1.f + lg - m * m - expf(lg));
  }
  s_z2[d] = z2;
  s_ent[d] = ent;
  s_act[d] = (kls * (1.f / 64.f) > 0.01f) ? 1 : 0;
  __syncthreads();
  for (int st = 128; st > 0; st >>= 1) {
    if (d < st) {
      s_z2[d] += s_z2[d + st];
      s_ent[d] += s_ent[d + st];
      s_act[d] += s_act[d + st];
    }
    __syncthreads();
  }
  if (d == 0) {
    float sse = 0.f;
    for (int i = 0; i < npart; ++i) sse += part[i];
    const float L2PI = 1.8378770664093453f;            // log(2*pi)
    float prior = -0.5f * s_z2[0] - 0.5f * 16384.f * L2PI;
    float entropy = 0.5f * s_ent[0];
    out3[0] = -sse + prior + entropy;                  // elbo
    out3[1] = -prior - entropy;
    out3[2] = (float)s_act[0];                         // num_active_dims
  }
}

// ---------------------------------------------------------------------------
// Host-side orchestration
// ---------------------------------------------------------------------------
extern "C" void kernel_launch(void* const* d_in, const int* in_sizes, int n_in,
                              void* d_out, int out_size, void* d_ws, size_t ws_size,
                              hipStream_t stream) {
  (void)in_sizes; (void)n_in; (void)out_size; (void)ws_size;

  const float* x    = (const float*)d_in[0];
  const float* e    = (const float*)d_in[1];
  const float* ew1  = (const float*)d_in[2];
  const float* eb1  = (const float*)d_in[3];
  const float* ew2  = (const float*)d_in[4];
  const float* eb2  = (const float*)d_in[5];
  const float* ew3  = (const float*)d_in[6];
  const float* eb3  = (const float*)d_in[7];
  const float* rew1 = (const float*)d_in[8];
  const float* reb1 = (const float*)d_in[9];
  const float* rew2 = (const float*)d_in[10];
  const float* reb2 = (const float*)d_in[11];
  const float* wmu  = (const float*)d_in[12];
  const float* bmu  = (const float*)d_in[13];
  const float* wlv  = (const float*)d_in[14];
  const float* blv  = (const float*)d_in[15];
  const float* wd   = (const float*)d_in[16];
  const float* bd   = (const float*)d_in[17];
  const float* rdw1 = (const float*)d_in[18];
  const float* rdb1 = (const float*)d_in[19];
  const float* rdw2 = (const float*)d_in[20];
  const float* rdb2 = (const float*)d_in[21];
  const float* dw1  = (const float*)d_in[22];
  const float* db1  = (const float*)d_in[23];
  const float* dw2  = (const float*)d_in[24];
  const float* db2  = (const float*)d_in[25];
  float* out = (float*)d_out;

  // ---- workspace arena (256B aligned slots, recomputed identically each call)
  size_t off = 0;
  auto alloc = [&](size_t bytes) -> void* {
    void* p = (char*)d_ws + off;
    off += (bytes + 255) & ~(size_t)255;
    return p;
  };
  __bf16* ew2b  = (__bf16*)alloc(131072 * 2);      // [128][1024]
  __bf16* ew3b  = (__bf16*)alloc(147456 * 2);      // [128][1152]
  __bf16* rew1b = (__bf16*)alloc(147456 * 2);      // [2][64][1152]
  __bf16* rew2b = (__bf16*)alloc(16384 * 2);       // [2][128][64]
  __bf16* rdw1b = (__bf16*)alloc(147456 * 2);
  __bf16* rdw2b = (__bf16*)alloc(16384 * 2);
  __bf16* dw1p  = (__bf16*)alloc(131072 * 2);      // [4][64][512]
  __bf16* wmut  = (__bf16*)alloc((size_t)8388608 * 2);  // [256][32768]
  __bf16* wlvt  = (__bf16*)alloc((size_t)8388608 * 2);
  __bf16* wdt   = (__bf16*)alloc((size_t)8388608 * 2);  // [32768][256]
  __bf16* flatb = (__bf16*)alloc((size_t)2097152 * 2);  // [64][32768]
  __bf16* zbf   = (__bf16*)alloc(16384 * 2);
  float*  muB   = (float*)alloc(16384 * 4);
  float*  lvB   = (float*)alloc(16384 * 4);
  float*  zF    = (float*)alloc(16384 * 4);
  float*  h1    = (float*)alloc((size_t)4194304 * 4);   // [64,64,32,32]; reused as gup
  float*  h2    = (float*)alloc((size_t)2097152 * 4);   // [64,128,16,16]
  float*  h3    = (float*)alloc((size_t)2097152 * 4);
  float*  g     = (float*)alloc((size_t)2097152 * 4);
  float*  tbuf  = (float*)alloc((size_t)1048576 * 4);   // [64,64,16,16]
  __bf16* col   = (__bf16*)alloc((size_t)18874368 * 2); // max patch matrix 16384x1152
  float*  part  = (float*)alloc(512 * 4);
  float*  gup   = h1;   // h1 dead after conv2's im2col

  auto NB = [](long total, int t) { return (unsigned)((total + t - 1) / t); };

  auto gemm = [&](const __bf16* A, const __bf16* Wt, const float* bias,
                  const float* resid, float* o, int M, int N, int K,
                  int P, int WA, int ys, int xs, int istr, int relu) {
    unsigned blocks = (unsigned)((M / 64) * (N / 64));
    k_gemm_bf16_wmma<<<blocks, 128, 0, stream>>>(A, Wt, bias, resid, o,
                                                 M, N, K, P, WA, ys, xs, istr, relu);
  };
  auto im2col = [&](const float* in, __bf16* c, int C, int H, int W, int OH, int OW,
                    int KH, int KW, int s, int pady, int padx, int relu) {
    long total = (long)64 * OH * OW * C * KH * KW;
    k_im2col<<<NB(total, 256), 256, 0, stream>>>(in, c, C, H, W, OH, OW,
                                                 KH, KW, s, pady, padx, relu, total);
  };

  // ---- weight preparation (bf16)
  k_cast_bf16<<<NB(131072, 256), 256, 0, stream>>>(ew2, ew2b, 131072);
  k_cast_bf16<<<NB(147456, 256), 256, 0, stream>>>(ew3, ew3b, 147456);
  k_cast_bf16<<<NB(147456, 256), 256, 0, stream>>>(rew1, rew1b, 147456);
  k_cast_bf16<<<NB(16384, 256), 256, 0, stream>>>(rew2, rew2b, 16384);
  k_cast_bf16<<<NB(147456, 256), 256, 0, stream>>>(rdw1, rdw1b, 147456);
  k_cast_bf16<<<NB(16384, 256), 256, 0, stream>>>(rdw2, rdw2b, 16384);
  k_cast_tr_bf16<<<NB(8388608, 256), 256, 0, stream>>>(wmu, wmut, 32768, 256);
  k_cast_tr_bf16<<<NB(8388608, 256), 256, 0, stream>>>(wlv, wlvt, 32768, 256);
  k_cast_tr_bf16<<<NB(8388608, 256), 256, 0, stream>>>(wd, wdt, 256, 32768);
  k_dw1_par_cast<<<NB(131072, 256), 256, 0, stream>>>(dw1, dw1p);

  // ---- encoder
  k_conv1<<<NB(4194304, 256), 256, 0, stream>>>(x, ew1, eb1, h1);

  im2col(h1, col, 64, 32, 32, 16, 16, 4, 4, 2, 1, 1, 0);
  gemm(col, ew2b, eb2, nullptr, h2, 16384, 128, 1024, 256, 16, 16, 1, 256, 1);

  im2col(h2, col, 128, 16, 16, 16, 16, 3, 3, 1, 1, 1, 0);
  gemm(col, ew3b, eb3, nullptr, h3, 16384, 128, 1152, 256, 16, 16, 1, 256, 0);

  for (int i = 0; i < 2; ++i) {   // encoder residual stack
    im2col(h3, col, 128, 16, 16, 16, 16, 3, 3, 1, 1, 1, 1);
    gemm(col, rew1b + i * 73728, reb1 + i * 64, nullptr, tbuf,
         16384, 64, 1152, 256, 16, 16, 1, 256, 0);
    im2col(tbuf, col, 64, 16, 16, 16, 16, 1, 1, 1, 0, 0, 1);
    gemm(col, rew2b + i * 8192, reb2 + i * 128, h3, h3,
         16384, 128, 64, 256, 16, 16, 1, 256, 0);
  }
  k_cast_relu_bf16<<<NB(2097152, 256), 256, 0, stream>>>(h3, flatb, 2097152);

  // ---- latent heads + tree reparameterization
  gemm(flatb, wmut, bmu, nullptr, muB, 64, 256, 32768, 0, 0, 0, 0, 0, 0);
  gemm(flatb, wlvt, blv, nullptr, lvB, 64, 256, 32768, 0, 0, 0, 0, 0, 0);
  k_reparam<<<NB(16384, 256), 256, 0, stream>>>(muB, lvB, e, zF, zbf);

  // ---- decoder
  gemm(zbf, wdt, bd, nullptr, g, 64, 32768, 256, 0, 0, 0, 0, 0, 0);

  for (int i = 0; i < 2; ++i) {   // decoder residual stack
    im2col(g, col, 128, 16, 16, 16, 16, 3, 3, 1, 1, 1, 1);
    gemm(col, rdw1b + i * 73728, rdb1 + i * 64, nullptr, tbuf,
         16384, 64, 1152, 256, 16, 16, 1, 256, 0);
    im2col(tbuf, col, 64, 16, 16, 16, 16, 1, 1, 1, 0, 0, 1);
    gemm(col, rdw2b + i * 8192, rdb2 + i * 128, g, g,
         16384, 128, 64, 256, 16, 16, 1, 256, 0);
  }

  // convT1 as 4 output-parity GEMMs (K = 128*4 valid taps); res-stack ReLU folded
  for (int par = 0; par < 4; ++par) {
    int py = par >> 1, px = par & 1;
    im2col(g, col, 128, 16, 16, 16, 16, 2, 2, 1, 1 - py, 1 - px, 1);
    gemm(col, dw1p + par * 32768, db1, nullptr, gup + py * 32 + px,
         16384, 64, 512, 256, 16, 64, 2, 1024, 1);
  }

  k_convT2<<<NB(786432, 256), 256, 0, stream>>>(gup, dw2, db2, out);

  // ---- losses
  k_sse_partial<<<512, 256, 0, stream>>>(out, x, part, 786432);
  k_finalize<<<1, 256, 0, stream>>>(muB, lvB, zF, part, 512, out + 786432);
}
